// LGAGenerator_86045374808276
// MI455X (gfx1250) — compile-verified, matrix operand order checked by reading, not run on hardware
//
#include <hip/hip_runtime.h>
#include <hip/hip_bf16.h>

typedef __attribute__((ext_vector_type(16))) _Float16 v16h;
typedef __attribute__((ext_vector_type(8)))  _Float16 v8h;
typedef __attribute__((ext_vector_type(8)))  float    v8f;

#define NNODES 10000
#define NEDGES 320000
#define INDIM  128
#define HIDDIM 64
#define MTILES 625   // 10000 / 16

// ---------------- small elementwise / setup kernels ----------------

__global__ void k_deg_init(float* deg, int n) {
  int i = blockIdx.x * blockDim.x + threadIdx.x;
  if (i < n) deg[i] = 1.0f;                 // self-loop contributes 1 to degree
}

__global__ void k_deg_acc(const int* __restrict__ dst, float* deg, int e) {
  int i = blockIdx.x * blockDim.x + threadIdx.x;
  if (i < e) atomicAdd(&deg[dst[i]], 1.0f);
}

__global__ void k_rsqrt(float* d, int n) {
  int i = blockIdx.x * blockDim.x + threadIdx.x;
  if (i < n) d[i] = rsqrtf(d[i]);           // deg >= 1 always (self-loops)
}

__global__ void k_cvt_f32_f16(const float* __restrict__ x, _Float16* __restrict__ y, int n) {
  int i = blockIdx.x * blockDim.x + threadIdx.x;
  if (i < n) y[i] = (_Float16)x[i];
}

// W: K x 64 row-major (fp32)  ->  WT: 64 x K row-major (f16)
__global__ void k_w_transpose(const float* __restrict__ W, _Float16* __restrict__ WT, int K) {
  int idx = blockIdx.x * blockDim.x + threadIdx.x;
  if (idx >= K * HIDDIM) return;
  int k = idx / HIDDIM;
  int n = idx % HIDDIM;
  WT[(size_t)n * K + k] = (_Float16)W[idx];
}

// h[i*64+f] = dis[i]^2 * xw[i*64+f] + bias[f]   (self-loop message + bias)
__global__ void k_init_h(const float* __restrict__ xw, const float* __restrict__ dis,
                         const float* __restrict__ bias, float* __restrict__ h, int total) {
  int i = blockIdx.x * blockDim.x + threadIdx.x;
  if (i >= total) return;
  int node = i >> 6;
  int f = i & 63;
  float w = dis[node];
  h[i] = w * w * xw[i] + bias[f];
}

// one 64-lane group per edge: h[dst] += dis[src]*dis[dst] * xw[src]
__global__ void k_scatter(const int* __restrict__ src, const int* __restrict__ dst,
                          const float* __restrict__ dis, const float* __restrict__ xw,
                          float* __restrict__ h, int e) {
  int g = blockIdx.x * 4 + (threadIdx.x >> 6);
  int f = threadIdx.x & 63;
  if (g >= e) return;
  int s = src[g];
  int d = dst[g];
  float w = dis[s] * dis[d];
  atomicAdd(&h[(size_t)d * HIDDIM + f], w * xw[(size_t)s * HIDDIM + f]);
}

__global__ void k_relu(float* h, int n) {
  int i = blockIdx.x * blockDim.x + threadIdx.x;
  if (i < n) h[i] = fmaxf(h[i], 0.0f);
}

// ---------------- WMMA fragment loader (CDNA5 wave32 16-bit layout) ----------------
// A 16x32 f16 fragment, lane l: M = l&15; halves j=0..7 -> K = k0 + 8*(l>>4) + j,
// halves j=8..15 -> K = k0 + 16 + 8*(l>>4) + (j-8). Two contiguous 16B chunks.
__device__ __forceinline__ v16h load_frag(const _Float16* __restrict__ row, int k0, int hofs) {
  v8h lo = *(const v8h*)(row + k0 + hofs);
  v8h hi = *(const v8h*)(row + k0 + 16 + hofs);
  return __builtin_shufflevector(lo, hi, 0,1,2,3,4,5,6,7,8,9,10,11,12,13,14,15);
}

// ---------------- GEMM: C[M x 64] = A[M x K](f16) * BT[64 x K](f16)^T ----------------
// one wave per 16x16 output tile; 8 waves per block
__global__ void k_gemm_wmma(const _Float16* __restrict__ A, const _Float16* __restrict__ BT,
                            float* __restrict__ C, int K, int ntiles) {
  int wid  = (blockIdx.x * blockDim.x + threadIdx.x) >> 5;
  int lane = threadIdx.x & 31;
  if (wid >= ntiles) return;
  int mt = wid >> 2;                // 4 n-tiles (N=64)
  int nt = wid & 3;
  int l16  = lane & 15;
  int hofs = (lane >> 4) * 8;
  const _Float16* arow = A  + (size_t)(mt * 16 + l16) * K;
  const _Float16* brow = BT + (size_t)(nt * 16 + l16) * K;
  v8f c = {};
  for (int k0 = 0; k0 < K; k0 += 32) {
    v16h a = load_frag(arow, k0, hofs);
    v16h b = load_frag(brow, k0, hofs);
    c = __builtin_amdgcn_wmma_f32_16x16x32_f16(false, a, false, b, (short)0, c, false, false);
  }
  int rbase = mt * 16 + (lane >> 4) * 8;
  int col   = nt * 16 + l16;
#pragma unroll
  for (int r = 0; r < 8; r++)
    C[(size_t)(rbase + r) * HIDDIM + col] = c[r];
}

// ---------------- link prediction: P = sigmoid(H * H^T), H: 10000 x 64 f16 ----------------
// one wave per 16x16 tile of P; B tile = rows of H (H^T columns), same loader as A.
__global__ void k_linkpred(const _Float16* __restrict__ H, float* __restrict__ P) {
  int lane = threadIdx.x & 31;
  int w    = threadIdx.x >> 5;
  int nt   = blockIdx.x;                 // 0..624
  int mt   = blockIdx.y * 8 + w;
  if (mt >= MTILES) return;
  int l16  = lane & 15;
  int hofs = (lane >> 4) * 8;
  const _Float16* arow = H + (size_t)(mt * 16 + l16) * HIDDIM;
  const _Float16* brow = H + (size_t)(nt * 16 + l16) * HIDDIM;
  v8f c = {};
#pragma unroll
  for (int k0 = 0; k0 < HIDDIM; k0 += 32) {
    v16h a = load_frag(arow, k0, hofs);
    v16h b = load_frag(brow, k0, hofs);
    c = __builtin_amdgcn_wmma_f32_16x16x32_f16(false, a, false, b, (short)0, c, false, false);
  }
  int rbase = mt * 16 + (lane >> 4) * 8;
  int col   = nt * 16 + l16;
#pragma unroll
  for (int r = 0; r < 8; r++) {
    float s = 1.0f / (1.0f + __expf(-c[r]));
    P[(size_t)(rbase + r) * NNODES + col] = s;
  }
}

// ---------------- host-side orchestration ----------------

static inline size_t alignup(size_t x) { return (x + 255) & ~(size_t)255; }

extern "C" void kernel_launch(void* const* d_in, const int* in_sizes, int n_in,
                              void* d_out, int out_size, void* d_ws, size_t ws_size,
                              hipStream_t stream) {
  const float* x  = (const float*)d_in[0];
  const int*   ei = (const int*)d_in[1];
  const float* W1 = (const float*)d_in[2];
  const float* b1 = (const float*)d_in[3];
  const float* W2 = (const float*)d_in[4];
  const float* b2 = (const float*)d_in[5];
  float* out = (float*)d_out;

  const int* src = ei;            // edge_index[0]
  const int* dst = ei + NEDGES;   // edge_index[1]

  // workspace carve-up
  char* ws = (char*)d_ws;
  size_t off = 0;
  float*    dis = (float*)(ws + off);    off += alignup((size_t)NNODES * 4);
  _Float16* xh  = (_Float16*)(ws + off); off += alignup((size_t)NNODES * INDIM * 2);
  _Float16* w1t = (_Float16*)(ws + off); off += alignup((size_t)HIDDIM * INDIM * 2);
  _Float16* w2t = (_Float16*)(ws + off); off += alignup((size_t)HIDDIM * HIDDIM * 2);
  float*    xw  = (float*)(ws + off);    off += alignup((size_t)NNODES * HIDDIM * 4);
  float*    h   = (float*)(ws + off);    off += alignup((size_t)NNODES * HIDDIM * 4);
  _Float16* hh  = (_Float16*)(ws + off); off += alignup((size_t)NNODES * HIDDIM * 2);
  (void)ws_size; (void)n_in; (void)in_sizes; (void)out_size;

  const int B = 256;
  const int nh = NNODES * HIDDIM;
  const int ntiles = MTILES * 4;                       // 2500 tiles per layer GEMM
  const int gemm_blocks = (ntiles * 32 + B - 1) / B;   // 8 waves / block

  // degree -> deg_inv_sqrt
  k_deg_init<<<(NNODES + B - 1) / B, B, 0, stream>>>(dis, NNODES);
  k_deg_acc<<<(NEDGES + B - 1) / B, B, 0, stream>>>(dst, dis, NEDGES);
  k_rsqrt<<<(NNODES + B - 1) / B, B, 0, stream>>>(dis, NNODES);

  // f16 staging of X, W1^T, W2^T
  k_cvt_f32_f16<<<(NNODES * INDIM + B - 1) / B, B, 0, stream>>>(x, xh, NNODES * INDIM);
  k_w_transpose<<<(INDIM * HIDDIM + B - 1) / B, B, 0, stream>>>(W1, w1t, INDIM);
  k_w_transpose<<<(HIDDIM * HIDDIM + B - 1) / B, B, 0, stream>>>(W2, w2t, HIDDIM);

  // layer 1: XW1 -> normalize/aggregate -> +b1 -> relu
  k_gemm_wmma<<<gemm_blocks, B, 0, stream>>>(xh, w1t, xw, INDIM, ntiles);
  k_init_h<<<(nh + B - 1) / B, B, 0, stream>>>(xw, dis, b1, h, nh);
  k_scatter<<<(NEDGES + 3) / 4, B, 0, stream>>>(src, dst, dis, xw, h, NEDGES);
  k_relu<<<(nh + B - 1) / B, B, 0, stream>>>(h, nh);

  // layer 2: HW2 -> normalize/aggregate -> +b2
  k_cvt_f32_f16<<<(nh + B - 1) / B, B, 0, stream>>>(h, hh, nh);
  k_gemm_wmma<<<gemm_blocks, B, 0, stream>>>(hh, w2t, xw, HIDDIM, ntiles);
  k_init_h<<<(nh + B - 1) / B, B, 0, stream>>>(xw, dis, b2, h, nh);
  k_scatter<<<(NEDGES + 3) / 4, B, 0, stream>>>(src, dst, dis, xw, h, NEDGES);

  // link prediction: sigmoid(H H^T)
  k_cvt_f32_f16<<<(nh + B - 1) / B, B, 0, stream>>>(h, hh, nh);
  dim3 grid(MTILES, (MTILES + 7) / 8);
  k_linkpred<<<grid, B, 0, stream>>>(hh, out);
}
